// InteractionLayerNView_75350906241383
// MI455X (gfx1250) — matrix operand order repaired and checked by reading, not compiled, not run on hardware
//
#include <hip/hip_runtime.h>
#include <hip/hip_bf16.h>
#include <cstdint>

typedef __attribute__((ext_vector_type(16))) __bf16 v16bf;
typedef __attribute__((ext_vector_type(8)))  __bf16 bf16x8;
typedef __attribute__((ext_vector_type(4)))  __bf16 bf16x4;
typedef __attribute__((ext_vector_type(8)))  float  v8f;

#define D_MODEL   1024
#define NHEADS    16
#define HEAD_DIM  64
#define TOKS      16            // tokens per workgroup
#define ROWS      48            // TOKS * 3 views
#define SA        1040          // bf16 row stride for sA / sCtx (pad 16)
#define SB        40            // bf16 row stride for staged B-tiles (32 K + pad 8)
#define SQ        196           // fp32 row stride for per-head qkv buffer
#define QCOLS     192           // q|k|v head columns (3 * 64)
#define KCHUNKS   32            // 1024 / 32

#define B1_TILE   (QCOLS * 32)  // 6144 bf16 per (head, kchunk) tile
#define B2_TILE   (128 * 32)    // 4096 bf16 per (nchunk, kchunk) tile
#define WS_B1_ELEMS ((size_t)NHEADS * KCHUNKS * B1_TILE)   // 3,145,728
#define WS_B2_ELEMS ((size_t)8 * KCHUNKS * B2_TILE)        // 1,048,576

// LDS: sA + sCtx (2*99840) + 2x sBt (30720) + sQKV (37632) + sS (768) = 268800 B
#define SMEM_BYTES (2u*ROWS*SA*2u + 2u*QCOLS*SB*2u + ROWS*SQ*4u + 16u*12u*4u)

__device__ __forceinline__ v8f zero8() {
  v8f z;
#pragma unroll
  for (int j = 0; j < 8; ++j) z[j] = 0.0f;
  return z;
}

__device__ __forceinline__ bf16x4 pack4(float4 f) {
  bf16x4 o;
  o[0] = (__bf16)f.x; o[1] = (__bf16)f.y;
  o[2] = (__bf16)f.z; o[3] = (__bf16)f.w;
  return o;
}

// A fragment (16-bit 16x32): lane<16 holds K pairs {0..7,16..23}, lane>=16 {8..15,24..31}
__device__ __forceinline__ v16bf load_a_frag(const __bf16* row_kbase, int loA) {
  v16bf a;
  float4* ap = reinterpret_cast<float4*>(&a);
  const float4* sp = reinterpret_cast<const float4*>(row_kbase + loA);
  ap[0] = sp[0];   // K = loA .. loA+7
  ap[1] = sp[2];   // K = loA+16 .. loA+23
  return a;
}

// B fragment (32x16, K-major per column): lane<16 holds K {0..7,16..23}, lane>=16 {8..15,24..31}
__device__ __forceinline__ v16bf load_b_frag(const __bf16* col_base, int loB) {
  v16bf b;
  float4* bp = reinterpret_cast<float4*>(&b);
  const float4* sp = reinterpret_cast<const float4*>(col_base + loB);
  bp[0] = sp[0];
  bp[1] = sp[2];
  return b;
}

// Pure b128 copy of a pre-converted, pre-tiled bf16 weight tile into padded LDS.
__device__ __forceinline__ void stage_tile(const __bf16* __restrict__ src,
                                           __bf16* __restrict__ dst,
                                           int tid, int nchunks_of_256) {
  // src: [n][32] contiguous; dst: [n][SB] padded; 8 bf16 (16B) per copy
  for (int p = 0; p < nchunks_of_256; ++p) {
    int i = tid + p * 256;
    int n = i >> 2, k8 = (i & 3) << 3;
    *reinterpret_cast<bf16x8*>(dst + n * SB + k8) =
        *reinterpret_cast<const bf16x8*>(src + n * 32 + k8);
  }
}

// ---- One-time weight conversion fp32 -> bf16, pre-tiled for the staging loops ----
// wsB1[((h*32+kk)*6144) + n*32 + k] = in_proj_w[c(n,h)][kk*32+k]
// wsB2[((nch*32+kk)*4096) + n*32 + k] = out_proj_w[nch*128+n][kk*32+k]
__global__ __launch_bounds__(256)
void convert_w_kernel(const float* __restrict__ inW, const float* __restrict__ outW,
                      uint16_t* __restrict__ wsB1raw, uint16_t* __restrict__ wsB2raw)
{
  __bf16* wsB1 = reinterpret_cast<__bf16*>(wsB1raw);
  __bf16* wsB2 = reinterpret_cast<__bf16*>(wsB2raw);
  size_t idx4 = ((size_t)blockIdx.x * 256 + threadIdx.x) * 4;
  if (idx4 < WS_B1_ELEMS) {
    size_t rem = idx4;
    int hh = (int)(rem / (KCHUNKS * B1_TILE)); rem %= (KCHUNKS * B1_TILE);
    int kk = (int)(rem / B1_TILE);
    int j  = (int)(rem % B1_TILE);
    int n = j >> 5, k = j & 31;
    int c = ((n >> 6) << 10) + hh * HEAD_DIM + (n & 63);
    float4 f = *reinterpret_cast<const float4*>(inW + (size_t)c * D_MODEL + kk * 32 + k);
    *reinterpret_cast<bf16x4*>(wsB1 + idx4) = pack4(f);
  } else if (idx4 < WS_B1_ELEMS + WS_B2_ELEMS) {
    size_t rem = idx4 - WS_B1_ELEMS;
    int nch = (int)(rem / (KCHUNKS * B2_TILE));
    size_t r2 = rem % (KCHUNKS * B2_TILE);
    int kk = (int)(r2 / B2_TILE);
    int j  = (int)(r2 % B2_TILE);
    int n = j >> 5, k = j & 31;
    int c = nch * 128 + n;
    float4 f = *reinterpret_cast<const float4*>(outW + (size_t)c * D_MODEL + kk * 32 + k);
    *reinterpret_cast<bf16x4*>(wsB2 + (idx4 - WS_B1_ELEMS)) = pack4(f);
  }
}

__global__ __launch_bounds__(256)
void fused_mha_kernel(const float* __restrict__ h0, const float* __restrict__ h1,
                      const float* __restrict__ h2,
                      const uint16_t* __restrict__ wsB1raw,
                      const uint16_t* __restrict__ wsB2raw,
                      const float* __restrict__ inB, const float* __restrict__ outB,
                      float* __restrict__ xout)
{
  const __bf16* wsB1 = reinterpret_cast<const __bf16*>(wsB1raw);
  const __bf16* wsB2 = reinterpret_cast<const __bf16*>(wsB2raw);

  extern __shared__ char smemraw[];
  __bf16* sA   = reinterpret_cast<__bf16*>(smemraw);        // [ROWS][SA]  input h (bf16)
  __bf16* sCtx = sA + ROWS * SA;                            // [ROWS][SA]  attention ctx (bf16)
  __bf16* sBt0 = sCtx + ROWS * SA;                          // double-buffered B tiles
  __bf16* sBt1 = sBt0 + QCOLS * SB;
  float*  sQKV = reinterpret_cast<float*>(sBt1 + QCOLS * SB);// [ROWS][SQ] per-head qkv
  float*  sS   = sQKV + ROWS * SQ;                          // [16][12]    scores / attn

  const int tid   = threadIdx.x;
  const int wave  = tid >> 5;
  const int lane  = tid & 31;
  const int lhalf = lane & 15;
  const int hi    = lane >> 4;
  const int loA   = hi << 3;   // A-frag K offset (0/8)
  const int loB   = hi << 3;   // B-frag K offset (0/8)
  const int hi8   = hi << 3;   // C/D row offset (0/8)

  const size_t tok0 = (size_t)blockIdx.x * TOKS;

  // ---- Phase 0: stage h (16 tokens x 3 views x 1024) as bf16 A-matrix ----
  for (int i = tid; i < ROWS * (D_MODEL / 4); i += 256) {
    int r  = i >> 8;
    int c4 = (i & 255) << 2;
    int tb = r / 3, vw = r % 3;
    const float* hsrc = (vw == 0) ? h0 : ((vw == 1) ? h1 : h2);
    const float4 f = *reinterpret_cast<const float4*>(hsrc + (tok0 + tb) * D_MODEL + c4);
    *reinterpret_cast<bf16x4*>(sA + r * SA + c4) = pack4(f);
  }

  // Static GEMM1 tile ownership: 36 tiles (3 M x 12 N); wave w owns ids {w, w+8, ...}.
  // Waves with id >= 36 redundantly recompute tile 0 (no branches around WMMA; EXEC
  // stays all-ones as the ISA requires); their stores are masked in the epilogue.
  bool g1_act[5];
  int  g1_col[5];                 // N column this lane covers (0..191)
  int  g1_boff[5];                // staged-B column offset (buffer-relative)
  const __bf16* g1_a[5];          // A-matrix row base (per lane)
#pragma unroll
  for (int t = 0; t < 5; ++t) {
    int id = wave + t * 8;
    g1_act[t] = (id < 36);
    int idc = g1_act[t] ? id : 0;
    int mt = idc % 3, nt = idc / 3;
    g1_a[t]    = sA + (mt * 16 + lhalf) * SA;
    g1_boff[t] = (nt * 16 + lhalf) * SB;
    g1_col[t]  = nt * 16 + lhalf;
  }
  __syncthreads();

  // ---- Per-head: QKV GEMM -> tiny attention -> ctx (bf16) ----
  for (int h = 0; h < NHEADS; ++h) {
    const __bf16* wsrc = wsB1 + (size_t)h * KCHUNKS * B1_TILE;
    v8f acc[5];
#pragma unroll
    for (int t = 0; t < 5; ++t) acc[t] = zero8();

    stage_tile(wsrc, sBt0, tid, 3);          // prologue: tile kk=0
    __syncthreads();
    for (int kk = 0; kk < KCHUNKS; ++kk) {
      __bf16* cur = (kk & 1) ? sBt1 : sBt0;
      __bf16* nxt = (kk & 1) ? sBt0 : sBt1;
      if (kk + 1 < KCHUNKS) {                // stage next tile while computing
        const __bf16* s = wsrc + (size_t)(kk + 1) * B1_TILE;
        if (kk + 2 < KCHUNKS) __builtin_prefetch(s + B1_TILE + tid * 24, 0, 1);
        stage_tile(s, nxt, tid, 3);
      }
      const int kb = kk * 32;
      v16bf af[5], bfr[5];
#pragma unroll
      for (int t = 0; t < 5; ++t) {          // issue all fragment loads first
        af[t]  = load_a_frag(g1_a[t] + kb, loA);
        bfr[t] = load_b_frag(cur + g1_boff[t], loB);
      }
#pragma unroll
      for (int t = 0; t < 5; ++t) {          // then an uninterrupted WMMA burst
        acc[t] = __builtin_amdgcn_wmma_f32_16x16x32_bf16(
            false, af[t], false, bfr[t], (short)0, acc[t], false, false);
      }
      __syncthreads();                       // one barrier per K-step
    }
    // epilogue: bias add, spill head qkv to LDS fp32 (stores masked for dup tiles)
#pragma unroll
    for (int t = 0; t < 5; ++t) {
      if (g1_act[t]) {
        int id = wave + t * 8;
        int mt = id % 3;
        int col = g1_col[t];
        int c = ((col >> 6) << 10) + h * HEAD_DIM + (col & 63);
        float bias = inB[c];
#pragma unroll
        for (int j = 0; j < 8; ++j) {
          int row = mt * 16 + hi8 + j;
          sQKV[row * SQ + col] = acc[t][j] + bias;
        }
      }
    }
    __syncthreads();

    // ---- tiny 3x3 attention per token (16 threads per token) ----
    {
      int tok = tid >> 4, sub = tid & 15;
      if (sub < 9) {
        int vv = sub / 3, ww = sub % 3;
        const float4* q4 = reinterpret_cast<const float4*>(sQKV + (tok * 3 + vv) * SQ);
        const float4* k4 = reinterpret_cast<const float4*>(sQKV + (tok * 3 + ww) * SQ + HEAD_DIM);
        float s = 0.f;
#pragma unroll
        for (int d = 0; d < HEAD_DIM / 4; ++d) {
          float4 qa = q4[d], kb4 = k4[d];
          s += qa.x * kb4.x + qa.y * kb4.y + qa.z * kb4.z + qa.w * kb4.w;
        }
        sS[tok * 12 + sub] = s * 0.125f;   // 1/sqrt(64)
      }
      __syncthreads();
      if (sub < 3) {
        float s0 = sS[tok * 12 + sub * 3 + 0];
        float s1 = sS[tok * 12 + sub * 3 + 1];
        float s2 = sS[tok * 12 + sub * 3 + 2];
        float m  = fmaxf(s0, fmaxf(s1, s2));
        float e0 = __expf(s0 - m), e1 = __expf(s1 - m), e2 = __expf(s2 - m);
        float inv = 1.f / (e0 + e1 + e2);
        sS[tok * 12 + sub * 3 + 0] = e0 * inv;
        sS[tok * 12 + sub * 3 + 1] = e1 * inv;
        sS[tok * 12 + sub * 3 + 2] = e2 * inv;
      }
      __syncthreads();
      {
        int dd = sub * 4;                    // each thread owns 4 consecutive head dims
        float4 v0 = *reinterpret_cast<const float4*>(sQKV + (tok * 3 + 0) * SQ + 128 + dd);
        float4 v1 = *reinterpret_cast<const float4*>(sQKV + (tok * 3 + 1) * SQ + 128 + dd);
        float4 v2 = *reinterpret_cast<const float4*>(sQKV + (tok * 3 + 2) * SQ + 128 + dd);
#pragma unroll
        for (int vv = 0; vv < 3; ++vv) {
          float a0 = sS[tok * 12 + vv * 3 + 0];
          float a1 = sS[tok * 12 + vv * 3 + 1];
          float a2 = sS[tok * 12 + vv * 3 + 2];
          bf16x4 o;
          o[0] = (__bf16)(a0 * v0.x + a1 * v1.x + a2 * v2.x);
          o[1] = (__bf16)(a0 * v0.y + a1 * v1.y + a2 * v2.y);
          o[2] = (__bf16)(a0 * v0.z + a1 * v1.z + a2 * v2.z);
          o[3] = (__bf16)(a0 * v0.w + a1 * v1.w + a2 * v2.w);
          *reinterpret_cast<bf16x4*>(sCtx + (tok * 3 + vv) * SA + h * HEAD_DIM + dd) = o;
        }
      }
      __syncthreads();
    }
  }

  // ---- GEMM2: attn_out = ctx @ out_proj^T; + bias + residual -> xout (pre-LN) ----
  // 24 tiles per 128-col chunk = 3 M x 8 N over 8 waves; wave owns ids {w, w+8, w+16}.
  int g2_mt[3], g2_nt[3], g2_boff[3];
  const __bf16* g2_a[3];
#pragma unroll
  for (int t = 0; t < 3; ++t) {
    int id = wave + t * 8;
    g2_mt[t]   = id % 3;
    g2_nt[t]   = id / 3;
    g2_a[t]    = sCtx + (g2_mt[t] * 16 + lhalf) * SA;
    g2_boff[t] = (g2_nt[t] * 16 + lhalf) * SB;
  }

  for (int nch = 0; nch < 8; ++nch) {          // 128 output columns per chunk
    const __bf16* wsrc = wsB2 + (size_t)nch * KCHUNKS * B2_TILE;
    v8f acc2[3];
#pragma unroll
    for (int t = 0; t < 3; ++t) acc2[t] = zero8();

    stage_tile(wsrc, sBt0, tid, 2);
    __syncthreads();
    for (int kk = 0; kk < KCHUNKS; ++kk) {
      __bf16* cur = (kk & 1) ? sBt1 : sBt0;
      __bf16* nxt = (kk & 1) ? sBt0 : sBt1;
      if (kk + 1 < KCHUNKS) {
        const __bf16* s = wsrc + (size_t)(kk + 1) * B2_TILE;
        if (kk + 2 < KCHUNKS) __builtin_prefetch(s + B2_TILE + tid * 16, 0, 1);
        stage_tile(s, nxt, tid, 2);
      }
      const int kb = kk * 32;
      v16bf af[3], bfr[3];
#pragma unroll
      for (int t = 0; t < 3; ++t) {
        af[t]  = load_a_frag(g2_a[t] + kb, loA);
        bfr[t] = load_b_frag(cur + g2_boff[t], loB);
      }
#pragma unroll
      for (int t = 0; t < 3; ++t) {
        acc2[t] = __builtin_amdgcn_wmma_f32_16x16x32_bf16(
            false, af[t], false, bfr[t], (short)0, acc2[t], false, false);
      }
      __syncthreads();
    }
#pragma unroll
    for (int t = 0; t < 3; ++t) {
      int col = nch * 128 + g2_nt[t] * 16 + lhalf;
      float bias = outB[col];
#pragma unroll
      for (int j = 0; j < 8; ++j) {
        int row = g2_mt[t] * 16 + hi8 + j;
        int tb = row / 3, vw = row % 3;
        size_t gt = tok0 + tb;
        const float* hsrc = (vw == 0) ? h0 : ((vw == 1) ? h1 : h2);
        float x = acc2[t][j] + bias + hsrc[gt * D_MODEL + col];   // exact fp32 residual
        xout[(gt * 3 + vw) * D_MODEL + col] = x;
      }
    }
    __syncthreads();   // next nchunk's prologue staging reuses sBt0
  }
}

// ---- In-place row LayerNorm over 49152 rows of 1024 (bandwidth-bound) ----
__global__ __launch_bounds__(256)
void ln_kernel(float* __restrict__ x, const float* __restrict__ w, const float* __restrict__ b)
{
  __shared__ float red[18];
  const int tid = threadIdx.x;
  float* xr = x + (size_t)blockIdx.x * D_MODEL;
  float v[4]; float s = 0.f, ss = 0.f;
#pragma unroll
  for (int j = 0; j < 4; ++j) {
    v[j] = xr[tid + j * 256];
    s += v[j]; ss += v[j] * v[j];
  }
#pragma unroll
  for (int off = 16; off > 0; off >>= 1) {
    s  += __shfl_xor(s,  off, 32);
    ss += __shfl_xor(ss, off, 32);
  }
  if ((tid & 31) == 0) { red[tid >> 5] = s; red[8 + (tid >> 5)] = ss; }
  __syncthreads();
  if (tid == 0) {
    float S = 0.f, SS = 0.f;
    for (int q = 0; q < 8; ++q) { S += red[q]; SS += red[8 + q]; }
    float mu  = S * (1.f / D_MODEL);
    float var = SS * (1.f / D_MODEL) - mu * mu;
    red[16] = mu;
    red[17] = rsqrtf(var + 1e-5f);
  }
  __syncthreads();
  float mu = red[16], inv = red[17];
#pragma unroll
  for (int j = 0; j < 4; ++j) {
    int c = tid + j * 256;
    xr[c] = (v[j] - mu) * inv * w[c] + b[c];
  }
}

extern "C" void kernel_launch(void* const* d_in, const int* in_sizes, int n_in,
                              void* d_out, int out_size, void* d_ws, size_t ws_size,
                              hipStream_t stream)
{
  (void)in_sizes; (void)n_in; (void)ws_size; (void)out_size;
  const float* h0   = (const float*)d_in[0];
  const float* h1   = (const float*)d_in[1];
  const float* h2   = (const float*)d_in[2];
  const float* inW  = (const float*)d_in[3];
  const float* inB  = (const float*)d_in[4];
  const float* outW = (const float*)d_in[5];
  const float* outB = (const float*)d_in[6];
  const float* lnW  = (const float*)d_in[7];
  const float* lnB  = (const float*)d_in[8];
  float* xout = (float*)d_out;

  uint16_t* wsB1 = (uint16_t*)d_ws;                 // 6 MB bf16 tiled in_proj_w
  uint16_t* wsB2 = wsB1 + WS_B1_ELEMS;              // 2 MB bf16 tiled out_proj_w

  const int T = 16 * 1024;                          // tokens
  const size_t cvt_elems = WS_B1_ELEMS + WS_B2_ELEMS;
  const int cvt_blocks = (int)(cvt_elems / 4 / 256);  // 4096
  convert_w_kernel<<<cvt_blocks, 256, 0, stream>>>(inW, outW, wsB1, wsB2);

  const int nblocks = T / TOKS;                     // 1024 workgroups
  fused_mha_kernel<<<nblocks, 256, SMEM_BYTES, stream>>>(
      h0, h1, h2, wsB1, wsB2, inB, outB, xout);
  ln_kernel<<<T * 3, 256, 0, stream>>>(xout, lnW, lnB);
}
